// NearestNeighborSampler_12017318494554
// MI455X (gfx1250) — matrix operand order repaired and checked by reading, not compiled, not run on hardware
//
#include <hip/hip_runtime.h>

typedef _Float16 v8h  __attribute__((ext_vector_type(8)));
typedef _Float16 v16h __attribute__((ext_vector_type(16)));
typedef float    v8f  __attribute__((ext_vector_type(8)));

#define NQ 32768   // rows in data_queue
#define NA 1024    // rows in data
#define D  256     // feature dim
#define CT 16      // 16x16 col-tiles per wave per block

// Monotonic float -> uint mapping so integer max == float max.
__device__ __forceinline__ unsigned int ord_f32(float s) {
    unsigned int u = __float_as_uint(s);
    return (u & 0x80000000u) ? ~u : (u | 0x80000000u);
}

// 16 contiguous halves -> v16h (two 16B loads).
__device__ __forceinline__ v16h load_b16(const _Float16* p) {
    const v8h* q = (const v8h*)p;
    return __builtin_shufflevector(q[0], q[1],
                                   0,1,2,3,4,5,6,7,8,9,10,11,12,13,14,15);
}

// A 16-bit 16x32 fragment: lane holds K = {0..7, 16..23} + 8*g of the chunk.
__device__ __forceinline__ v16h load_a16(const _Float16* p) {
    const v8h* q = (const v8h*)p;
    return __builtin_shufflevector(q[0], q[2],
                                   0,1,2,3,4,5,6,7,8,9,10,11,12,13,14,15);
}

__device__ __forceinline__ v8f wmma_f16(v16h a, v16h b, v8f c) {
    return __builtin_amdgcn_wmma_f32_16x16x32_f16(false, a, false, b,
                                                  (short)0, c, false, false);
}

// ---------------------------------------------------------------------------
// Kernel 1: split data_queue rows into f16 hi/lo, compute q_sq, reset best[].
// One block (256 threads) per queue row.
// ---------------------------------------------------------------------------
__global__ void nns_prep_kernel(const float* __restrict__ data,
                                const float* __restrict__ queue,
                                _Float16* __restrict__ Qhi,
                                _Float16* __restrict__ Qlo,
                                float* __restrict__ qsq,
                                unsigned long long* __restrict__ best) {
    const int j   = blockIdx.x;
    const int tid = threadIdx.x;
    const float* src = (j < NA) ? (data + (size_t)j * D)
                                : (queue + (size_t)(j - NA) * D);
    float x = src[tid];
    _Float16 h = (_Float16)x;
    _Float16 l = (_Float16)(x - (float)h);
    Qhi[(size_t)j * D + tid] = h;
    Qlo[(size_t)j * D + tid] = l;

    float v = x * x;
    #pragma unroll
    for (int off = 16; off > 0; off >>= 1) v += __shfl_xor(v, off, 32);

    __shared__ float wsum[8];
    const int lane = tid & 31, wid = tid >> 5;
    if (lane == 0) wsum[wid] = v;
    __syncthreads();
    if (tid == 0) {
        float s = 0.f;
        #pragma unroll
        for (int w = 0; w < 8; ++w) s += wsum[w];
        qsq[j] = s;
        if (j < NA) best[j] = 0ull;  // re-init every launch (graph replay safe)
    }
}

// ---------------------------------------------------------------------------
// Kernel 2: split-precision f16 WMMA GEMM + fused argmax.
// Grid: (NA/16 row tiles, NQ/2048 col strips). Block: 256 threads = 8 waves.
// dot = ah*bh + ah*bl + al*bh with 3 independent accumulator chains, and the
// B fragments double-buffered one k-chunk ahead so global_load latency
// overlaps the WMMAs instead of forcing full s_wait_loadcnt drains.
// ---------------------------------------------------------------------------
__global__ void __launch_bounds__(256)
nns_gemm_argmax_kernel(const float* __restrict__ data,
                       const _Float16* __restrict__ Qhi,
                       const _Float16* __restrict__ Qlo,
                       const float* __restrict__ qsq,
                       unsigned long long* __restrict__ best) {
    __shared__ _Float16 Ahi[16][D];
    __shared__ _Float16 Alo[16][D];

    const int tid  = threadIdx.x;
    const int lane = tid & 31;
    const int wave = tid >> 5;
    const int g    = lane >> 4;   // half-wave index (0/1)
    const int hl   = lane & 15;   // lane within half-wave
    const int row_base = blockIdx.x * 16;

    // Stage + hi/lo split the 16x256 A tile into LDS (coalesced, full f32 read).
    for (int e = tid; e < 16 * D; e += 256) {
        const int r = e >> 8;          // D == 256
        const int c = e & (D - 1);
        float x = data[(size_t)(row_base + r) * D + c];
        _Float16 h = (_Float16)x;
        Ahi[r][c] = h;
        Alo[r][c] = (_Float16)(x - (float)h);
    }
    __syncthreads();

    unsigned long long bk[8];
    #pragma unroll
    for (int r = 0; r < 8; ++r) bk[r] = 0ull;

    for (int t = 0; t < CT; ++t) {
        const int col_base = blockIdx.y * (8 * 16 * CT) + t * 128 + wave * 16;
        // Lane hl supplies column (col_base+hl); half-wave g supplies
        // K = 16*g .. 16*g+15 of each 32-chunk (contiguous k order, same as A).
        const _Float16* qh = Qhi + (size_t)(col_base + hl) * D + 16 * g;
        const _Float16* ql = Qlo + (size_t)(col_base + hl) * D + 16 * g;
        if (t + 1 < CT) {  // near-scope prefetch of next tile's B rows
            __builtin_prefetch(qh + (size_t)128 * D, 0, 3);
            __builtin_prefetch(ql + (size_t)128 * D, 0, 3);
        }

        v8f acc_hh = {};   // ah * bh
        v8f acc_hl = {};   // ah * bl
        v8f acc_lh = {};   // al * bh

        // Software-pipelined B double buffer (one k-chunk lookahead).
        v16h bh_buf[2], bl_buf[2];
        bh_buf[0] = load_b16(qh);
        bl_buf[0] = load_b16(ql);

        #pragma unroll
        for (int kk = 0; kk < 8; ++kk) {
            const int cur = kk & 1;
            if (kk < 7) {
                bh_buf[cur ^ 1] = load_b16(qh + (kk + 1) * 32);
                bl_buf[cur ^ 1] = load_b16(ql + (kk + 1) * 32);
            }
            v16h ah = load_b16(&Ahi[hl][kk * 32 + 16 * g]);
            v16h al = load_b16(&Alo[hl][kk * 32 + 16 * g]);
            acc_hh = wmma_f16(ah, bh_buf[cur], acc_hh);
            acc_hl = wmma_f16(ah, bl_buf[cur], acc_hl);
            acc_lh = wmma_f16(al, bh_buf[cur], acc_lh);
        }

        // C layout: element r is row M = r + 8*g, col N = hl.
        const int j = col_base + hl;
        const float qs = qsq[j];
        const unsigned int jkey = ~(unsigned int)j;  // ties resolve to smaller j
        #pragma unroll
        for (int r = 0; r < 8; ++r) {
            const int i = row_base + r + 8 * g;
            const float dot = acc_hh[r] + acc_hl[r] + acc_lh[r];
            const float score = 2.0f * dot - qs;
            if (j != i) {  // diagonal -> -inf (simply excluded)
                unsigned long long key =
                    ((unsigned long long)ord_f32(score) << 32) | (unsigned long long)jkey;
                if (key > bk[r]) bk[r] = key;
            }
        }
    }

    // Reduce over the 16 columns held by each half-wave, then one atomic per row.
    #pragma unroll
    for (int r = 0; r < 8; ++r) {
        unsigned long long v = bk[r];
        #pragma unroll
        for (int off = 1; off < 16; off <<= 1) {
            unsigned long long o = __shfl_xor(v, off, 32);
            if (o > v) v = o;
        }
        if (hl == 0) atomicMax(&best[row_base + r + 8 * g], v);
    }
}

// ---------------------------------------------------------------------------
// Kernel 3: gather winning rows. One block (64 threads) per output row.
// ---------------------------------------------------------------------------
__global__ void nns_gather_kernel(const float* __restrict__ data,
                                  const float* __restrict__ queue,
                                  const unsigned long long* __restrict__ best,
                                  float* __restrict__ out) {
    const int i = blockIdx.x;
    unsigned int j = ~(unsigned int)(best[i] & 0xFFFFFFFFull);
    if (j >= NQ) j = 0;
    const float* src = (j < NA) ? (data + (size_t)j * D)
                                : (queue + (size_t)(j - NA) * D);
    const float4* s4 = (const float4*)src;
    float4* o4 = (float4*)(out + (size_t)i * D);
    o4[threadIdx.x] = s4[threadIdx.x];   // 64 threads x float4 = 256 floats
}

extern "C" void kernel_launch(void* const* d_in, const int* in_sizes, int n_in,
                              void* d_out, int out_size, void* d_ws, size_t ws_size,
                              hipStream_t stream) {
    const float* data  = (const float*)d_in[0];   // (1024, 256) f32
    const float* queue = (const float*)d_in[1];   // (32768, 256) f32
    float* out = (float*)d_out;                   // (1024, 256) f32

    // Workspace carve-up (~33.7 MB total).
    _Float16* Qhi = (_Float16*)d_ws;                               // 16 MB
    _Float16* Qlo = Qhi + (size_t)NQ * D;                          // 16 MB
    float*    qsq = (float*)(Qlo + (size_t)NQ * D);                // 128 KB
    unsigned long long* best = (unsigned long long*)(qsq + NQ);    // 8 KB

    nns_prep_kernel<<<NQ, D, 0, stream>>>(data, queue, Qhi, Qlo, qsq, best);
    nns_gemm_argmax_kernel<<<dim3(NA / 16, NQ / (8 * 16 * CT)), 256, 0, stream>>>(
        data, Qhi, Qlo, qsq, best);
    nns_gather_kernel<<<NA, 64, 0, stream>>>(data, queue, best, out);
}
